// Writer_7679401525561
// MI455X (gfx1250) — compile-verified, matrix operand order checked by reading, not run on hardware
//
#include <hip/hip_runtime.h>
#include <math.h>

// Problem sizes (match reference).
#define BB   128
#define TT   2048
#define DD   512
#define TWOD 1024

typedef __attribute__((ext_vector_type(16))) _Float16 v16h;
typedef __attribute__((ext_vector_type(8)))  float    v8f;
typedef __attribute__((ext_vector_type(4)))  float    f4;

union Frag16 {
    v16h v;
    _Float16 h[16];
    unsigned int u[8];
};

// CDNA5 WMMA 16x16x32 f16: A element i (0..15) in a lane maps to
// k = (i<8 ? i : i+8) + 8*(lane>=16); row/col = lane & 15.
__device__ __forceinline__ int kmap(int i, int hi) {
    return (i < 8 ? i : i + 8) + hi * 8;
}

// ---------------------------------------------------------------------------
// Kernel 0: fold W_ca into control (cw[b,d] = control[b,d]*W_ca[d]), zero msa.
// ---------------------------------------------------------------------------
__global__ __launch_bounds__(256) void prep_kernel(
    const float* __restrict__ control, const float* __restrict__ W_ca,
    float* __restrict__ cw, float* __restrict__ msa)
{
    int i = blockIdx.x * 256 + threadIdx.x;      // 0 .. B*D-1
    cw[i]  = control[i] * W_ca[i & (DD - 1)];
    msa[i] = 0.0f;
}

// ---------------------------------------------------------------------------
// Kernel 1: attention logits  ca[b,t] = dot(past_controls[b,t,:], cw[b,:]) + b_ca
// One wave32 per (b,t) row; 4x b128 non-temporal loads per lane; shfl reduce.
// ---------------------------------------------------------------------------
__global__ __launch_bounds__(256) void logits_kernel(
    const float* __restrict__ past, const float* __restrict__ cw,
    const float* __restrict__ b_ca, float* __restrict__ ca)
{
    int g    = (blockIdx.x * 256 + threadIdx.x) >> 5;   // global wave id = b*T + t
    int lane = threadIdx.x & 31;
    int b    = g >> 11;                                  // T = 2048
    int t    = g & (TT - 1);

    const f4* pc = (const f4*)(past + ((size_t)(b * TT + t)) * TWOD); // controls half
    const f4* c4 = (const f4*)(cw + b * DD);

    float s = 0.0f;
#pragma unroll
    for (int i = 0; i < 4; ++i) {
        f4 a = __builtin_nontemporal_load(pc + lane + 32 * i); // streamed once: NT
        f4 w = c4[lane + 32 * i];                              // L2-resident
        s += a.x * w.x + a.y * w.y + a.z * w.z + a.w * w.w;
    }
#pragma unroll
    for (int off = 16; off >= 1; off >>= 1) s += __shfl_xor(s, off, 32);

    if (lane == 0) {
        float v = s + b_ca[0];
        ca[g] = (v == 0.0f) ? -1e9f : v;   // exact-zero padding mask, as reference
    }
}

// ---------------------------------------------------------------------------
// Kernel 2: softmax over T per batch, in place. 256 threads, 8 elems/thread.
// ---------------------------------------------------------------------------
__global__ __launch_bounds__(256) void softmax_kernel(float* __restrict__ ca)
{
    __shared__ float red[8];
    __shared__ float bc;
    float* row = ca + (size_t)blockIdx.x * TT;
    int tid = threadIdx.x;

    float v[8];
    float mx = -3.4e38f;
#pragma unroll
    for (int i = 0; i < 8; ++i) { v[i] = row[tid + 256 * i]; mx = fmaxf(mx, v[i]); }
#pragma unroll
    for (int off = 16; off >= 1; off >>= 1) mx = fmaxf(mx, __shfl_xor(mx, off, 32));
    if ((tid & 31) == 0) red[tid >> 5] = mx;
    __syncthreads();
    if (tid == 0) {
        float m = red[0];
        for (int i = 1; i < 8; ++i) m = fmaxf(m, red[i]);
        bc = m;
    }
    __syncthreads();
    mx = bc;

    float s = 0.0f;
#pragma unroll
    for (int i = 0; i < 8; ++i) { v[i] = __expf(v[i] - mx); s += v[i]; }
#pragma unroll
    for (int off = 16; off >= 1; off >>= 1) s += __shfl_xor(s, off, 32);
    if ((tid & 31) == 0) red[tid >> 5] = s;
    __syncthreads();
    if (tid == 0) {
        float tsum = 0.0f;
        for (int i = 0; i < 8; ++i) tsum += red[i];
        bc = tsum;
    }
    __syncthreads();
    float inv = 1.0f / bc;
#pragma unroll
    for (int i = 0; i < 8; ++i) row[tid + 256 * i] = v[i] * inv;
}

// ---------------------------------------------------------------------------
// Kernel 3: msa[b,:] = sum_t w[b,t] * past_memories[b,t,:]
// Block = (b, T-chunk of 512); 128 threads, one f4 of d per thread; NT b128
// streaming loads; atomicAdd accumulate across the 4 chunks.
// ---------------------------------------------------------------------------
__global__ __launch_bounds__(128) void msa_kernel(
    const float* __restrict__ past, const float* __restrict__ w,
    float* __restrict__ msa)
{
    int b = blockIdx.x >> 2, chunk = blockIdx.x & 3;
    int d4 = threadIdx.x * 4;
    const float* base = past + ((size_t)(b * TT) + (size_t)chunk * 512) * TWOD + DD + d4;
    const float* wr   = w + b * TT + chunk * 512;

    f4 acc = (f4){0.0f, 0.0f, 0.0f, 0.0f};
#pragma unroll 4
    for (int t = 0; t < 512; ++t) {
        float wt = wr[t];
        f4 m = __builtin_nontemporal_load((const f4*)(base + (size_t)t * TWOD));
        acc.x += wt * m.x; acc.y += wt * m.y; acc.z += wt * m.z; acc.w += wt * m.w;
    }
    float* o = msa + b * DD + d4;
    atomicAdd(o + 0, acc.x); atomicAdd(o + 1, acc.y);
    atomicAdd(o + 2, acc.z); atomicAdd(o + 3, acc.w);
}

// ---------------------------------------------------------------------------
// Kernel 4 (WMMA tail):
//   m1 = [memory||read] @ W_m1 + b_m1                (16x1024 @ 1024x512 per block)
//   mp = m1 @ W_m2 + b_m2 + msa @ W_s + b_s          (two chained 16x512x512)
//   out = sigmoid(control@W_m3+b_m3) blend with memory
// 8 blocks x 256 threads (8 waves); wave owns 4 N-tiles of 16 cols.
// ---------------------------------------------------------------------------
__global__ __launch_bounds__(256) void tail_kernel(
    const float* __restrict__ memory, const float* __restrict__ readv,
    const float* __restrict__ control,
    const float* __restrict__ W_m1, const float* __restrict__ b_m1,
    const float* __restrict__ W_m2, const float* __restrict__ b_m2,
    const float* __restrict__ W_s,  const float* __restrict__ b_s,
    const float* __restrict__ W_m3, const float* __restrict__ b_m3,
    const float* __restrict__ msa,  float* __restrict__ out)
{
    __shared__ _Float16 sX[16 * 1032];   // X tile f16, padded row stride (bank-safe)
    __shared__ float    sM1[16 * 516];   // m1 tile f32, padded
    __shared__ float    sGate[16];

    const int tid  = threadIdx.x;
    const int wid  = tid >> 5;
    const int lane = tid & 31;
    const int hi   = lane >> 4;
    const int lo   = lane & 15;
    const int bm   = blockIdx.x * 16;

    // Stage X = [memory || read] tile as f16 (coalesced).
#pragma unroll
    for (int j = 0; j < 64; ++j) {
        int idx = tid + 256 * j;                 // 16*1024 elements
        int m = idx >> 10, c = idx & 1023;
        float v = (c < DD) ? memory[(bm + m) * DD + c]
                           : readv[(bm + m) * DD + (c - DD)];
        sX[m * 1032 + c] = (_Float16)v;
    }

    // Sigmoid gate per row: wave wid computes rows 2*wid, 2*wid+1.
#pragma unroll
    for (int r = 0; r < 2; ++r) {
        int m = wid * 2 + r;
        const float* cr = control + (bm + m) * DD;
        float s = 0.0f;
#pragma unroll
        for (int i = 0; i < 16; ++i) {
            int d = lane + 32 * i;
            s += cr[d] * W_m3[d];
        }
#pragma unroll
        for (int off = 16; off >= 1; off >>= 1) s += __shfl_xor(s, off, 32);
        if (lane == 0) sGate[m] = 1.0f / (1.0f + __expf(-(s + b_m3[0])));
    }
    __syncthreads();

    // ---- GEMM1: m1 = X(16x1024) @ W_m1(1024x512) ----
    v8f acc[4];
#pragma unroll
    for (int nt = 0; nt < 4; ++nt) acc[nt] = (v8f){0,0,0,0,0,0,0,0};

    const unsigned int* sXd = (const unsigned int*)sX;   // dword view of f16 pairs
    for (int ks = 0; ks < 32; ++ks) {
        const int kbase = ks * 32;
        Frag16 a;
#pragma unroll
        for (int j = 0; j < 8; ++j) {
            int k0 = (j < 4 ? 2 * j : 2 * j + 8) + hi * 8 + kbase;   // even k
            a.u[j] = sXd[lo * 516 + (k0 >> 1)];
        }
#pragma unroll
        for (int nt = 0; nt < 4; ++nt) {
            int n = wid * 64 + nt * 16 + lo;
            Frag16 bf;
#pragma unroll
            for (int i = 0; i < 16; ++i) {
                int k = kmap(i, hi) + kbase;
                bf.h[i] = (_Float16)W_m1[k * DD + n];
            }
            acc[nt] = __builtin_amdgcn_wmma_f32_16x16x32_f16(
                false, a.v, false, bf.v, (short)0, acc[nt], false, false);
        }
    }

    // m1 tile (+b_m1) -> LDS f32. C/D layout: m = r + 8*hi, n = lo.
#pragma unroll
    for (int nt = 0; nt < 4; ++nt) {
        int n = wid * 64 + nt * 16 + lo;
#pragma unroll
        for (int r = 0; r < 8; ++r) {
            int m = r + hi * 8;
            sM1[m * 516 + n] = acc[nt][r] + b_m1[n];
        }
    }
    __syncthreads();

    // ---- GEMM2: mp = m1 @ W_m2 + msa_tile @ W_s (chained accumulators) ----
    v8f acc2[4];
#pragma unroll
    for (int nt = 0; nt < 4; ++nt) acc2[nt] = (v8f){0,0,0,0,0,0,0,0};

    for (int ks = 0; ks < 16; ++ks) {
        const int kbase = ks * 32;
        Frag16 a1, a2;
#pragma unroll
        for (int i = 0; i < 16; ++i) {
            int k = kmap(i, hi) + kbase;
            a1.h[i] = (_Float16)sM1[lo * 516 + k];
            a2.h[i] = (_Float16)msa[(bm + lo) * DD + k];
        }
#pragma unroll
        for (int nt = 0; nt < 4; ++nt) {
            int n = wid * 64 + nt * 16 + lo;
            Frag16 b1, b2;
#pragma unroll
            for (int i = 0; i < 16; ++i) {
                int k = kmap(i, hi) + kbase;
                b1.h[i] = (_Float16)W_m2[k * DD + n];
                b2.h[i] = (_Float16)W_s[k * DD + n];
            }
            acc2[nt] = __builtin_amdgcn_wmma_f32_16x16x32_f16(
                false, a1.v, false, b1.v, (short)0, acc2[nt], false, false);
            acc2[nt] = __builtin_amdgcn_wmma_f32_16x16x32_f16(
                false, a2.v, false, b2.v, (short)0, acc2[nt], false, false);
        }
    }

    // Epilogue: biases, gate blend with memory, store.
#pragma unroll
    for (int nt = 0; nt < 4; ++nt) {
        int n = wid * 64 + nt * 16 + lo;
        float bias = b_m2[n] + b_s[n];
#pragma unroll
        for (int r = 0; r < 8; ++r) {
            int m   = r + hi * 8;
            float g = sGate[m];
            float mp  = acc2[nt][r] + bias;
            float mem = memory[(bm + m) * DD + n];
            out[(bm + m) * DD + n] = mp * g + (1.0f - g) * mem;
        }
    }
}

// ---------------------------------------------------------------------------
extern "C" void kernel_launch(void* const* d_in, const int* in_sizes, int n_in,
                              void* d_out, int out_size, void* d_ws, size_t ws_size,
                              hipStream_t stream)
{
    const float* memory  = (const float*)d_in[0];
    const float* readv   = (const float*)d_in[1];
    const float* control = (const float*)d_in[2];
    const float* past    = (const float*)d_in[3];
    const float* W_m1    = (const float*)d_in[4];
    const float* b_m1    = (const float*)d_in[5];
    const float* W_ca    = (const float*)d_in[6];
    const float* b_ca    = (const float*)d_in[7];
    const float* W_m2    = (const float*)d_in[8];
    const float* b_m2    = (const float*)d_in[9];
    const float* W_s     = (const float*)d_in[10];
    const float* b_s     = (const float*)d_in[11];
    const float* W_m3    = (const float*)d_in[12];
    const float* b_m3    = (const float*)d_in[13];
    float* out = (float*)d_out;

    float* ws  = (float*)d_ws;
    float* cw  = ws;                      // B*D      = 65536 floats
    float* ca  = ws + 65536;              // B*T      = 262144 floats (reused as weights)
    float* msa = ws + 65536 + 262144;     // B*D      = 65536 floats

    prep_kernel   <<<256,             256, 0, stream>>>(control, W_ca, cw, msa);
    logits_kernel <<<(BB * TT) / 8,   256, 0, stream>>>(past, cw, b_ca, ca);
    softmax_kernel<<<BB,              256, 0, stream>>>(ca);
    msa_kernel    <<<BB * 4,          128, 0, stream>>>(past, ca, msa);
    tail_kernel   <<<BB / 16,         256, 0, stream>>>(memory, readv, control,
                                                        W_m1, b_m1, W_m2, b_m2,
                                                        W_s, b_s, W_m3, b_m3,
                                                        msa, out);
}
